// FoundationScarletPhotometryHead_24807731102287
// MI455X (gfx1250) — matrix operand order repaired and checked by reading, not compiled, not run on hardware
//
#include <hip/hip_runtime.h>
#include <hip/hip_bf16.h>
#include <math.h>

typedef _Float16 h16;
typedef __attribute__((ext_vector_type(16))) _Float16 v16h;
typedef __attribute__((ext_vector_type(8)))  _Float16 v8h;
typedef __attribute__((ext_vector_type(8)))  float    v8f;

static constexpr int    NSRC      = 1024;
static constexpr size_t DELTA_OFF = (size_t)NSRC * 961;      // 984064
static constexpr size_t FEAT_OFF  = 2 * DELTA_OFF;           // 1968128

// workspace offsets in h16 elements (all multiples of 16 -> 32B aligned frags)
static constexpr size_t WS_PW1T = 0;                         // 256x1024
static constexpr size_t WS_PW2T = WS_PW1T + 256 * 1024;      // 1024x256
static constexpr size_t WS_BNT  = WS_PW2T + 1024 * 256;      // 256x128
static constexpr size_t WS_TR1T = WS_BNT  + 256 * 128;       // 192x256
static constexpr size_t WS_TR2T = WS_TR1T + 192 * 256;       // 256x256
static constexpr size_t WS_MDT  = WS_TR2T + 256 * 256;       // 256x976
static constexpr size_t WS_ST1T = WS_MDT  + 256 * 976;       // 9 taps * 64x64
static constexpr size_t WS_ST2T = WS_ST1T + 9 * 64 * 64;

// ---------------- WMMA helpers ----------------
__device__ __forceinline__ v8f vzero8() {
  v8f z;
#pragma unroll
  for (int i = 0; i < 8; ++i) z[i] = 0.f;
  return z;
}

__device__ __forceinline__ v8f wmma_f16(v16h a, v16h b, v8f c) {
  return __builtin_amdgcn_wmma_f32_16x16x32_f16(false, a, false, b, (short)0, c, false, false);
}

// A-fragment (16x32 f16) from row-major [M][K] LDS buffer.
// lane 0-15: K 0-7 & 16-23 ; lane 16-31: K 8-15 & 24-31 (per ISA layout).
__device__ __forceinline__ v16h afrag_lds(const h16* A, int row, int kstride, int kt, int lane) {
  const h16* p = A + row * kstride + kt * 32 + ((lane >> 4) << 3);
  v8h lo = *(const v8h*)p;
  v8h hi = *(const v8h*)(p + 16);
  v16h a;
#pragma unroll
  for (int i = 0; i < 8; ++i) { a[i] = lo[i]; a[i + 8] = hi[i]; }
  return a;
}

// B-fragment from pre-swizzled fragment-linear global buffer: one 32B load.
__device__ __forceinline__ v16h bfrag_g(const h16* __restrict__ B, int kt, int ntiles, int nt, int lane) {
  return *(const v16h*)(B + (((size_t)(kt * ntiles + nt) * 32 + lane) << 4));
}

__device__ __forceinline__ float gelu_f(float x) {
  return 0.5f * x * (1.f + erff(x * 0.70710678118654752f));
}

// ---------------- weight swizzle: B[k][n] = src[n*rs + k*cs + off] ----------------
__global__ __launch_bounds__(256) void k_swizzleB(const float* __restrict__ src, h16* __restrict__ dst,
                                                  int K, int Np, int Nv, int rs, int cs, int off) {
  int idx = blockIdx.x * 256 + threadIdx.x;
  if (idx >= K * Np) return;
  int k = idx / Np, n = idx % Np;
  float v = (n < Nv) ? src[(size_t)n * rs + (size_t)k * cs + off] : 0.f;
  int kt = k >> 5, kk = k & 31, nt = n >> 4;
  int lane = (((kk >> 3) & 1) << 4) | (n & 15);
  int i = (kk & 7) | ((kk >> 4) << 3);
  int ntiles = Np >> 4;
  dst[(((size_t)(kt * ntiles + nt) * 32 + lane) << 4) + i] = (h16)v;
}

// ---------------- bottleneck branch: extract + ConvNeXt + proj + gauss pool ----------------
// LDS map (bytes), exactly 160KB so two workgroups fit one 320KB WGP:
//  xwin 0..65536 (128x256 h16; pad rows 121..127 reused for metadata)
//  Abuf 65536..131072, Pbuf 131072..163840 (128x128 h16)
//  metadata inside xwin pad rows: gauss@61952 (128 f32), bnvec@62464 (128 f32),
//  xi0@62976 yi0@63040 fwx@63104 fwy@63168
__global__ __launch_bounds__(256) void k_bn(
    const float* __restrict__ bottleneck, const float* __restrict__ pos,
    const float* __restrict__ dw_w, const float* __restrict__ dw_b,
    const float* __restrict__ ln_g, const float* __restrict__ ln_b,
    const float* __restrict__ pw1_b, const float* __restrict__ pw2_b,
    const float* __restrict__ gamma, const float* __restrict__ bn_b,
    const h16* __restrict__ pw1T, const h16* __restrict__ pw2T, const h16* __restrict__ bnT,
    float* __restrict__ out) {
  extern __shared__ char smem[];
  h16*   xwin  = (h16*)(smem);
  h16*   Abuf  = (h16*)(smem + 65536);
  h16*   Pbuf  = (h16*)(smem + 131072);
  float* gaussb = (float*)(smem + 61952);
  float* bnvec  = (float*)(smem + 62464);
  int*   xi0 = (int*)(smem + 62976);
  int*   yi0 = (int*)(smem + 63040);
  float* fwx = (float*)(smem + 63104);
  float* fwy = (float*)(smem + 63168);

  const int tid = threadIdx.x;
  const int n = blockIdx.x;
  const int lane = tid & 31;
  const int w = tid >> 5;

  // phase 0: zero Abuf (pad rows feed WMMA and must be 0), bnvec, coords, raw gauss
  {
    float* az = (float*)(smem + 65536);
    for (int i = tid; i < 16384; i += 256) az[i] = 0.f;
  }
  if (tid < 128) bnvec[tid] = 0.f;
  float px = pos[2 * n + 0], py = pos[2 * n + 1];
  float pbx = (px + 0.5f) * 0.25f - 0.5f;
  float pby = (py + 0.5f) * 0.25f - 0.5f;
  if (tid < 11) {
    float xs = fminf(fmaxf(pbx + (float)(tid - 5), 0.f), 255.f);
    int x0 = (int)floorf(xs); x0 = x0 < 0 ? 0 : (x0 > 254 ? 254 : x0);
    xi0[tid] = x0; fwx[tid] = xs - (float)x0;
    float ys = fminf(fmaxf(pby + (float)(tid - 5), 0.f), 255.f);
    int y0 = (int)floorf(ys); y0 = y0 < 0 ? 0 : (y0 > 254 ? 254 : y0);
    yi0[tid] = y0; fwy[tid] = ys - (float)y0;
  }
  if (tid < 121) {
    int gy = tid / 11, gx = tid % 11;
    float fy = (float)(gy - 5) / 5.f, fx = (float)(gx - 5) / 5.f;
    gaussb[tid] = expf(-(fx * fx + fy * fy) * 3.125f);
  }
  __syncthreads();

  if (tid == 0) {
    float s = 0.f;
    for (int i = 0; i < 121; ++i) s += gaussb[i];
    gaussb[127] = s;
  }
  // bilinear extraction: consecutive threads -> consecutive window pixels (rows < 121 only)
  for (int idx = tid; idx < 121 * 256; idx += 256) {
    int pix = idx % 121, c = idx / 121;
    int jy = pix / 11, jx = pix % 11;
    int x0 = xi0[jx], y0 = yi0[jy];
    float wx = fwx[jx], wy = fwy[jy];
    const float* f = bottleneck + (size_t)c * 65536;
    float v00 = f[y0 * 256 + x0],       v01 = f[y0 * 256 + x0 + 1];
    float v10 = f[(y0 + 1) * 256 + x0], v11 = f[(y0 + 1) * 256 + x0 + 1];
    float v = v00 * (1.f - wy) * (1.f - wx) + v01 * (1.f - wy) * wx
            + v10 * wy * (1.f - wx) + v11 * wy * wx;
    xwin[pix * 256 + c] = (h16)v;
  }
  __syncthreads();

  if (tid < 121) gaussb[tid] *= (1.f / gaussb[127]);

  // depthwise 7x7 (zero-padded window) + bias -> Abuf (m=pixel, k=channel)
  for (int idx = tid; idx < 121 * 256; idx += 256) {
    int pix = idx % 121, c = idx / 121;
    int pyy = pix / 11, pxx = pix % 11;
    float acc = 0.f;
    for (int ky = 0; ky < 7; ++ky) {
      int qy = pyy + ky - 3; if (qy < 0 || qy >= 11) continue;
      for (int kx = 0; kx < 7; ++kx) {
        int qx = pxx + kx - 3; if (qx < 0 || qx >= 11) continue;
        acc += (float)xwin[(qy * 11 + qx) * 256 + c] * dw_w[c * 49 + ky * 7 + kx];
      }
    }
    acc += dw_b[c];
    Abuf[pix * 256 + c] = (h16)acc;
  }
  __syncthreads();

  // LayerNorm over channels per pixel
  if (tid < 121) {
    float mu = 0.f;
    for (int c = 0; c < 256; ++c) mu += (float)Abuf[tid * 256 + c];
    mu *= (1.f / 256.f);
    float var = 0.f;
    for (int c = 0; c < 256; ++c) { float d = (float)Abuf[tid * 256 + c] - mu; var += d * d; }
    var *= (1.f / 256.f);
    float inv = rsqrtf(var + 1e-6f);
    for (int c = 0; c < 256; ++c) {
      float v = ((float)Abuf[tid * 256 + c] - mu) * inv * ln_g[c] + ln_b[c];
      Abuf[tid * 256 + c] = (h16)v;
    }
  }
  __syncthreads();

  // pw1 (256->1024) -> GELU -> pw2 (1024->256), k-tiled through LDS P buffer.
  v8f acc[8][2];
#pragma unroll
  for (int mt = 0; mt < 8; ++mt)
#pragma unroll
    for (int ci = 0; ci < 2; ++ci) acc[mt][ci] = vzero8();

  for (int jt = 0; jt < 8; ++jt) {
    // prefetch next j-tile's weight fragments (global_prefetch path)
    if (jt + 1 < 8) {
      __builtin_prefetch(pw1T + (((size_t)((jt + 1) * 8 + w) * 32 + lane) << 4), 0, 1);
      __builtin_prefetch(pw2T + (((size_t)((jt + 1) * 4 * 16 + (w << 1)) * 32 + lane) << 4), 0, 1);
    }
    // hoist pw1 B fragments: invariant across the M loop (8x reuse)
    v16h bw1[8];
#pragma unroll
    for (int kk = 0; kk < 8; ++kk) bw1[kk] = bfrag_g(pw1T, kk, 64, jt * 8 + w, lane);

    // P = gelu(A @ W1[:, jt*128 : jt*128+128] + b1) ; wave w owns local n-tile w
    {
      int col = jt * 128 + w * 16 + (lane & 15);
      float bias = pw1_b[col];
      int rbase = (lane >> 4) << 3;
      for (int mt = 0; mt < 8; ++mt) {
        v8f p = vzero8();
#pragma unroll
        for (int kk = 0; kk < 8; ++kk) {
          v16h a = afrag_lds(Abuf, mt * 16 + (lane & 15), 256, kk, lane);
          p = wmma_f16(a, bw1[kk], p);
        }
#pragma unroll
        for (int r = 0; r < 8; ++r) {
          Pbuf[(mt * 16 + r + rbase) * 128 + w * 16 + (lane & 15)] = (h16)gelu_f(p[r] + bias);
        }
      }
    }
    __syncthreads();

    // hoist pw2 B fragments: invariant across the M loop (8x reuse)
    v16h bw2[2][4];
#pragma unroll
    for (int ci = 0; ci < 2; ++ci)
#pragma unroll
      for (int kk = 0; kk < 4; ++kk)
        bw2[ci][kk] = bfrag_g(pw2T, jt * 4 + kk, 16, (w << 1) + ci, lane);

    // Out += P @ W2[jt*128 : jt*128+128, :]
#pragma unroll
    for (int mt = 0; mt < 8; ++mt)
#pragma unroll
      for (int ci = 0; ci < 2; ++ci)
#pragma unroll
        for (int kk = 0; kk < 4; ++kk) {
          v16h a = afrag_lds(Pbuf, mt * 16 + (lane & 15), 128, kk, lane);
          acc[mt][ci] = wmma_f16(a, bw2[ci][kk], acc[mt][ci]);
        }
    __syncthreads();
  }

  // residual + layer-scale -> H (reuse Abuf; pad rows get garbage, masked later)
#pragma unroll
  for (int mt = 0; mt < 8; ++mt)
#pragma unroll
    for (int ci = 0; ci < 2; ++ci) {
      int ct = (w << 1) + ci;
      int c = ct * 16 + (lane & 15);
      float gm = gamma[c], b2 = pw2_b[c];
#pragma unroll
      for (int r = 0; r < 8; ++r) {
        int m = mt * 16 + r + ((lane >> 4) << 3);
        float hv = (float)xwin[m * 256 + c] + (acc[mt][ci][r] + b2) * gm;
        Abuf[m * 256 + c] = (h16)hv;
      }
    }
  __syncthreads();

  // bn projection (256->128) + GELU + gaussian pooling (rows >= 121 masked out)
  {
    int o = w * 16 + (lane & 15);
    float bb = bn_b[o];
    v16h bwp[8];
#pragma unroll
    for (int kk = 0; kk < 8; ++kk) bwp[kk] = bfrag_g(bnT, kk, 8, w, lane);
    float colsum = 0.f;
    for (int mt = 0; mt < 8; ++mt) {
      v8f pa = vzero8();
#pragma unroll
      for (int kk = 0; kk < 8; ++kk) {
        v16h a = afrag_lds(Abuf, mt * 16 + (lane & 15), 256, kk, lane);
        pa = wmma_f16(a, bwp[kk], pa);
      }
#pragma unroll
      for (int r = 0; r < 8; ++r) {
        int m = mt * 16 + r + ((lane >> 4) << 3);
        if (m < 121) colsum += gaussb[m] * gelu_f(pa[r] + bb);
      }
    }
    atomicAdd(&bnvec[o], colsum);
  }
  __syncthreads();
  if (tid < 128) out[FEAT_OFF + (size_t)n * 192 + tid] = bnvec[tid];
}

// ---------------- stem branch: extract + 2x conv3x3 (implicit WMMA GEMM) + gauss pool ----------------
__device__ __forceinline__ v16h stem_afrag(const h16* win, int mt, int dy, int dx, int kk, int lane) {
  int m = mt * 16 + (lane & 15);
  v16h a;
  if (m < 961) {
    int pyy = m / 31, pxx = m % 31;
    const h16* p = win + ((pyy + dy) * 33 + (pxx + dx)) * 64 + kk * 32 + ((lane >> 4) << 3);
    v8h lo = *(const v8h*)p;
    v8h hi = *(const v8h*)(p + 16);
#pragma unroll
    for (int i = 0; i < 8; ++i) { a[i] = lo[i]; a[i + 8] = hi[i]; }
  } else {
#pragma unroll
    for (int i = 0; i < 16; ++i) a[i] = (h16)0.f;
  }
  return a;
}

// LDS map (bytes): W0 0..139392, W1 139392..278784, gauss 278784 (961 f32),
// stvec 282640 (64 f32), gsum 282896, coords 282912..283424
__global__ __launch_bounds__(256) void k_stem(
    const float* __restrict__ vis, const float* __restrict__ pos,
    const float* __restrict__ st1_b, const float* __restrict__ st2_b,
    const h16* __restrict__ st1T, const h16* __restrict__ st2T,
    float* __restrict__ out) {
  extern __shared__ char smem[];
  h16*   W0 = (h16*)(smem);
  h16*   W1 = (h16*)(smem + 139392);
  float* gaussb = (float*)(smem + 278784);
  float* stvec  = (float*)(smem + 282640);
  float* gsum   = (float*)(smem + 282896);
  int*   xi0 = (int*)(smem + 282912);
  int*   yi0 = (int*)(smem + 283040);
  float* fwx = (float*)(smem + 283168);
  float* fwy = (float*)(smem + 283296);

  const int tid = threadIdx.x, lane = tid & 31, w = tid >> 5;
  const int n = blockIdx.x;

  for (int i = tid; i < 69696; i += 256) ((float*)smem)[i] = 0.f;  // zero W0+W1
  if (tid < 64) stvec[tid] = 0.f;
  if (tid == 0) gsum[0] = 0.f;
  float px = pos[2 * n], py = pos[2 * n + 1];
  if (tid < 31) {
    float xs = fminf(fmaxf(px + (float)(tid - 15), 0.f), 1023.f);
    int x0 = (int)floorf(xs); x0 = x0 < 0 ? 0 : (x0 > 1022 ? 1022 : x0);
    xi0[tid] = x0; fwx[tid] = xs - (float)x0;
    float ys = fminf(fmaxf(py + (float)(tid - 15), 0.f), 1023.f);
    int y0 = (int)floorf(ys); y0 = y0 < 0 ? 0 : (y0 > 1022 ? 1022 : y0);
    yi0[tid] = y0; fwy[tid] = ys - (float)y0;
  }
  {
    float p = 0.f;
    for (int i = tid; i < 961; i += 256) {
      int gy = i / 31, gx = i % 31;
      float fy = (float)(gy - 15) / 15.f, fx = (float)(gx - 15) / 15.f;
      float g = expf(-(fx * fx + fy * fy) * 3.125f);
      gaussb[i] = g; p += g;
    }
    atomicAdd(gsum, p);
  }
  __syncthreads();
  {
    float inv = 1.f / gsum[0];
    for (int i = tid; i < 961; i += 256) gaussb[i] *= inv;
  }
  // bilinear extraction into padded 33x33x64 f16 window
  for (int idx = tid; idx < 961 * 64; idx += 256) {
    int pix = idx % 961, c = idx / 961;
    int jy = pix / 31, jx = pix % 31;
    int x0 = xi0[jx], y0 = yi0[jy];
    float wx = fwx[jx], wy = fwy[jy];
    const float* f = vis + (size_t)c * 1048576;
    float v00 = f[(size_t)y0 * 1024 + x0],       v01 = f[(size_t)y0 * 1024 + x0 + 1];
    float v10 = f[(size_t)(y0 + 1) * 1024 + x0], v11 = f[(size_t)(y0 + 1) * 1024 + x0 + 1];
    float v = v00 * (1.f - wy) * (1.f - wx) + v01 * (1.f - wy) * wx
            + v10 * wy * (1.f - wx) + v11 * wy * wx;
    W0[((jy + 1) * 33 + (jx + 1)) * 64 + c] = (h16)v;
  }
  __syncthreads();

  // per-wave fixed output-channel tile: weights hoisted once (30x reuse)
  const int nt  = w & 3;
  const int mlo = (w < 4) ? 0 : 31;
  const int mhi = (w < 4) ? 31 : 61;
  const int o   = nt * 16 + (lane & 15);

  // conv1: 9 shifted GEMMs, K=64, out -> gelu -> W1 (padded)
  {
    v16h bw[9][2];
#pragma unroll
    for (int tap = 0; tap < 9; ++tap)
#pragma unroll
      for (int kk = 0; kk < 2; ++kk) bw[tap][kk] = bfrag_g(st1T + tap * 4096, kk, 4, nt, lane);
    float bias = st1_b[o];
    for (int mt = mlo; mt < mhi; ++mt) {
      v8f acc = vzero8();
#pragma unroll
      for (int tap = 0; tap < 9; ++tap) {
        int dy = tap / 3, dx = tap % 3;
#pragma unroll
        for (int kk = 0; kk < 2; ++kk) {
          v16h a = stem_afrag(W0, mt, dy, dx, kk, lane);
          acc = wmma_f16(a, bw[tap][kk], acc);
        }
      }
#pragma unroll
      for (int r = 0; r < 8; ++r) {
        int m = mt * 16 + r + ((lane >> 4) << 3);
        if (m < 961) {
          int qy = m / 31, qx = m % 31;
          W1[((qy + 1) * 33 + (qx + 1)) * 64 + o] = (h16)gelu_f(acc[r] + bias);
        }
      }
    }
  }
  __syncthreads();

  // conv2 + fused gaussian pooling
  {
    v16h bw[9][2];
#pragma unroll
    for (int tap = 0; tap < 9; ++tap)
#pragma unroll
      for (int kk = 0; kk < 2; ++kk) bw[tap][kk] = bfrag_g(st2T + tap * 4096, kk, 4, nt, lane);
    float bias = st2_b[o];
    float s = 0.f;
    for (int mt = mlo; mt < mhi; ++mt) {
      v8f acc = vzero8();
#pragma unroll
      for (int tap = 0; tap < 9; ++tap) {
        int dy = tap / 3, dx = tap % 3;
#pragma unroll
        for (int kk = 0; kk < 2; ++kk) {
          v16h a = stem_afrag(W1, mt, dy, dx, kk, lane);
          acc = wmma_f16(a, bw[tap][kk], acc);
        }
      }
#pragma unroll
      for (int r = 0; r < 8; ++r) {
        int m = mt * 16 + r + ((lane >> 4) << 3);
        if (m < 961) s += gaussb[m] * gelu_f(acc[r] + bias);
      }
    }
    atomicAdd(&stvec[o], s);
  }
  __syncthreads();
  if (tid < 64) out[FEAT_OFF + (size_t)n * 192 + 128 + tid] = stvec[tid];
}

// ---------------- feat MLP: 192->256->256->961, tanh -> delta ----------------
// LDS: FA 0..49152 (128x192), H1 49152..114688 (128x256), H2 114688..180224
__global__ __launch_bounds__(256) void k_mlp(
    const float* __restrict__ tr1_b, const float* __restrict__ tr2_b, const float* __restrict__ md_b,
    const h16* __restrict__ tr1T, const h16* __restrict__ tr2T, const h16* __restrict__ mdT,
    float* __restrict__ out) {
  extern __shared__ char smem[];
  h16* FA = (h16*)(smem);
  h16* H1 = (h16*)(smem + 49152);
  h16* H2 = (h16*)(smem + 114688);
  const int tid = threadIdx.x, lane = tid & 31, w = tid >> 5;
  const int s0 = blockIdx.x * 128;

  for (int idx = tid; idx < 128 * 192; idx += 256) {
    int m = idx / 192, k = idx % 192;
    FA[idx] = (h16)out[FEAT_OFF + (size_t)(s0 + m) * 192 + k];
  }
  __syncthreads();

  // GEMM1: 128x192 @ 192x256
  {
    v16h bw[2][6];
#pragma unroll
    for (int ci = 0; ci < 2; ++ci)
#pragma unroll
      for (int kk = 0; kk < 6; ++kk) bw[ci][kk] = bfrag_g(tr1T, kk, 16, (w << 1) + ci, lane);
    for (int mt = 0; mt < 8; ++mt)
#pragma unroll
      for (int ci = 0; ci < 2; ++ci) {
        int ct = (w << 1) + ci;
        v8f acc = vzero8();
#pragma unroll
        for (int kk = 0; kk < 6; ++kk) {
          v16h a = afrag_lds(FA, mt * 16 + (lane & 15), 192, kk, lane);
          acc = wmma_f16(a, bw[ci][kk], acc);
        }
        int col = ct * 16 + (lane & 15);
        float bias = tr1_b[col];
#pragma unroll
        for (int r = 0; r < 8; ++r) {
          int m = mt * 16 + r + ((lane >> 4) << 3);
          H1[m * 256 + col] = (h16)gelu_f(acc[r] + bias);
        }
      }
  }
  __syncthreads();

  // GEMM2: 128x256 @ 256x256
  {
    v16h bw[2][8];
#pragma unroll
    for (int ci = 0; ci < 2; ++ci)
#pragma unroll
      for (int kk = 0; kk < 8; ++kk) bw[ci][kk] = bfrag_g(tr2T, kk, 16, (w << 1) + ci, lane);
    for (int mt = 0; mt < 8; ++mt)
#pragma unroll
      for (int ci = 0; ci < 2; ++ci) {
        int ct = (w << 1) + ci;
        v8f acc = vzero8();
#pragma unroll
        for (int kk = 0; kk < 8; ++kk) {
          v16h a = afrag_lds(H1, mt * 16 + (lane & 15), 256, kk, lane);
          acc = wmma_f16(a, bw[ci][kk], acc);
        }
        int col = ct * 16 + (lane & 15);
        float bias = tr2_b[col];
#pragma unroll
        for (int r = 0; r < 8; ++r) {
          int m = mt * 16 + r + ((lane >> 4) << 3);
          H2[m * 256 + col] = (h16)gelu_f(acc[r] + bias);
        }
      }
  }
  __syncthreads();

  // GEMM3: 128x256 @ 256x976(961 valid) -> tanh -> delta; nt outer for 8x weight reuse
  for (int nt = w; nt < 61; nt += 8) {
    v16h bw[8];
#pragma unroll
    for (int kk = 0; kk < 8; ++kk) bw[kk] = bfrag_g(mdT, kk, 61, nt, lane);
    int col = nt * 16 + (lane & 15);
    for (int mt = 0; mt < 8; ++mt) {
      v8f acc = vzero8();
#pragma unroll
      for (int kk = 0; kk < 8; ++kk) {
        v16h a = afrag_lds(H2, mt * 16 + (lane & 15), 256, kk, lane);
        acc = wmma_f16(a, bw[kk], acc);
      }
      if (col < 961) {
        float bias = md_b[col];
#pragma unroll
        for (int r = 0; r < 8; ++r) {
          int s = s0 + mt * 16 + r + ((lane >> 4) << 3);
          out[DELTA_OFF + (size_t)s * 961 + col] = tanhf(acc[r] + bias);
        }
      }
    }
  }
}

// ---------------- morphology: normalize / softplus-inverse / softplus / normalize ----------------
__global__ __launch_bounds__(256) void k_morph(const float* __restrict__ initm, float* __restrict__ out) {
  __shared__ float vals[961];
  __shared__ float rsum[2];
  const int tid = threadIdx.x, n = blockIdx.x;
  if (tid < 2) rsum[tid] = 0.f;
  __syncthreads();
  float p = 0.f;
  for (int i = tid; i < 961; i += 256) {
    float v = fmaxf(initm[(size_t)n * 961 + i], 0.f);
    vals[i] = v; p += v;
  }
  atomicAdd(&rsum[0], p);
  __syncthreads();
  float tot = rsum[0] + 1e-8f;
  float p2 = 0.f;
  for (int i = tid; i < 961; i += 256) {
    float im = vals[i] / tot;
    float x = fmaxf(im, 1e-8f);
    float base = x + logf(-expm1f(-x));
    float z = base + 1.5f * out[DELTA_OFF + (size_t)n * 961 + i];
    float sp = z > 20.f ? z : log1pf(expf(z));
    vals[i] = sp; p2 += sp;
  }
  atomicAdd(&rsum[1], p2);
  __syncthreads();
  float tot2 = rsum[1] + 1e-8f;
  for (int i = tid; i < 961; i += 256) out[(size_t)n * 961 + i] = vals[i] / tot2;
}

// ---------------- host ----------------
extern "C" void kernel_launch(void* const* d_in, const int* in_sizes, int n_in,
                              void* d_out, int out_size, void* d_ws, size_t ws_size,
                              hipStream_t stream) {
  (void)in_sizes; (void)n_in; (void)out_size; (void)ws_size;
  const float* bottleneck = (const float*)d_in[0];
  const float* vis        = (const float*)d_in[1];
  const float* pos        = (const float*)d_in[2];
  const float* init_m     = (const float*)d_in[3];
  const float* dw_w  = (const float*)d_in[4];
  const float* dw_b  = (const float*)d_in[5];
  const float* ln_g  = (const float*)d_in[6];
  const float* ln_b  = (const float*)d_in[7];
  const float* pw1_w = (const float*)d_in[8];
  const float* pw1_b = (const float*)d_in[9];
  const float* pw2_w = (const float*)d_in[10];
  const float* pw2_b = (const float*)d_in[11];
  const float* gamma = (const float*)d_in[12];
  const float* bn_w  = (const float*)d_in[13];
  const float* bn_b  = (const float*)d_in[14];
  const float* st1_w = (const float*)d_in[15];
  const float* st1_b = (const float*)d_in[16];
  const float* st2_w = (const float*)d_in[17];
  const float* st2_b = (const float*)d_in[18];
  const float* tr1_w = (const float*)d_in[19];
  const float* tr1_b = (const float*)d_in[20];
  const float* tr2_w = (const float*)d_in[21];
  const float* tr2_b = (const float*)d_in[22];
  const float* md_w  = (const float*)d_in[23];
  const float* md_b  = (const float*)d_in[24];
  float* out = (float*)d_out;
  h16* wsh = (h16*)d_ws;

  auto sw = [&](const float* src, h16* dst, int K, int Np, int Nv, int rs, int cs, int off) {
    int total = K * Np;
    k_swizzleB<<<(total + 255) / 256, 256, 0, stream>>>(src, dst, K, Np, Nv, rs, cs, off);
  };
  sw(pw1_w, wsh + WS_PW1T, 256, 1024, 1024, 256, 1, 0);
  sw(pw2_w, wsh + WS_PW2T, 1024, 256, 256, 1024, 1, 0);
  sw(bn_w,  wsh + WS_BNT,  256, 128, 128, 256, 1, 0);
  sw(tr1_w, wsh + WS_TR1T, 192, 256, 256, 192, 1, 0);
  sw(tr2_w, wsh + WS_TR2T, 256, 256, 256, 256, 1, 0);
  sw(md_w,  wsh + WS_MDT,  256, 976, 961, 256, 1, 0);
  for (int t = 0; t < 9; ++t) {
    sw(st1_w, wsh + WS_ST1T + (size_t)t * 4096, 64, 64, 64, 576, 9, t);
    sw(st2_w, wsh + WS_ST2T + (size_t)t * 4096, 64, 64, 64, 576, 9, t);
  }

  k_bn<<<NSRC, 256, 163840, stream>>>(bottleneck, pos, dw_w, dw_b, ln_g, ln_b,
                                      pw1_b, pw2_b, gamma, bn_b,
                                      wsh + WS_PW1T, wsh + WS_PW2T, wsh + WS_BNT, out);
  k_stem<<<NSRC, 256, 283424, stream>>>(vis, pos, st1_b, st2_b,
                                        wsh + WS_ST1T, wsh + WS_ST2T, out);
  k_mlp<<<NSRC / 128, 256, 180224, stream>>>(tr1_b, tr2_b, md_b,
                                             wsh + WS_TR1T, wsh + WS_TR2T, wsh + WS_MDT, out);
  k_morph<<<NSRC, 256, 0, stream>>>(init_m, out);
}
